// SelfAttention_10582799418056
// MI455X (gfx1250) — matrix operand order repaired and checked by reading, not compiled, not run on hardware
//
#include <hip/hip_runtime.h>
#include <hip/hip_bf16.h>

typedef __bf16 v16bf __attribute__((ext_vector_type(16)));
typedef __bf16 v8bf  __attribute__((ext_vector_type(8)));
typedef float  v8f   __attribute__((ext_vector_type(8)));

__device__ __forceinline__ v8f vzero8() {
  v8f z;
#pragma unroll
  for (int i = 0; i < 8; ++i) z[i] = 0.0f;
  return z;
}

__device__ __forceinline__ unsigned short f2bf(float f) {
  unsigned int u = __float_as_uint(f);
  u += 0x7FFFu + ((u >> 16) & 1u);   // round-to-nearest-even
  return (unsigned short)(u >> 16);
}

__device__ __forceinline__ v16bf cat8(v8bf lo, v8bf hi) {
  return __builtin_shufflevector(lo, hi, 0,1,2,3,4,5,6,7,8,9,10,11,12,13,14,15);
}

__device__ __forceinline__ v8f wmma_bf16(v16bf a, v16bf b, v8f c) {
  return __builtin_amdgcn_wmma_f32_16x16x32_bf16(false, a, false, b, (short)0, c,
                                                 false, false);
}

// ---------------------------------------------------------------------------
// Kernel 0: fp32 -> bf16 conversion
// ---------------------------------------------------------------------------
__global__ void cvt_bf16(const float* __restrict__ src,
                         unsigned short* __restrict__ dst, int n) {
  int i = blockIdx.x * blockDim.x + threadIdx.x;
  if (i < n) dst[i] = f2bf(src[i]);
}

// ---------------------------------------------------------------------------
// Kernel 1: QKV projection.  C[m,n] = sum_d x[m,d] * W[n,d] + b[n]
// Wave tile: 32(M) x 64(N)  -> 2 A-frags x 4 B-frags, 8 WMMAs per k-step,
// so each 32B B-operand load feeds two WMMAs (2x reuse vs round 1).
// Q,K stored row-major bf16 [16384,512]; V stored transposed per batch:
// VT[b][e][n] so PV B-fragments in the attention kernel are contiguous.
// ---------------------------------------------------------------------------
__global__ __launch_bounds__(128) void qkv_proj(
    const unsigned short* __restrict__ xb,
    const unsigned short* __restrict__ Wqb, const unsigned short* __restrict__ Wkb,
    const unsigned short* __restrict__ Wvb,
    const float* __restrict__ bq, const float* __restrict__ bk,
    const float* __restrict__ bv,
    unsigned short* __restrict__ Qb, unsigned short* __restrict__ Kb,
    unsigned short* __restrict__ VTb) {
  const int lane = threadIdx.x & 31;
  const int wave = threadIdx.x >> 5;
  const int half = lane >> 4;
  const int lm   = lane & 15;
  const int sel  = blockIdx.z;
  const int mtile = blockIdx.x * 128 + wave * 32;
  const int ntile = blockIdx.y * 64;

  const unsigned short* W = (sel == 0) ? Wqb : (sel == 1) ? Wkb : Wvb;
  const float* bias = (sel == 0) ? bq : (sel == 1) ? bk : bv;

  v8f acc[2][4];
#pragma unroll
  for (int i = 0; i < 2; ++i)
#pragma unroll
    for (int j = 0; j < 4; ++j) acc[i][j] = vzero8();

  const unsigned short* arow0 = xb + (size_t)(mtile + lm) * 512;
  const unsigned short* arow1 = arow0 + 16 * 512;
  const int k8 = half * 8;

#pragma unroll 4
  for (int t = 0; t < 16; ++t) {
    // A-fragments: two contiguous 16B chunks per lane.
    v16bf a0 = cat8(*(const v8bf*)(arow0 + t * 32 + 2 * k8),
                    *(const v8bf*)(arow0 + t * 32 + 2 * k8 + 8));
    v16bf a1 = cat8(*(const v8bf*)(arow1 + t * 32 + 2 * k8),
                    *(const v8bf*)(arow1 + t * 32 + 2 * k8 + 8));
#pragma unroll
    for (int j = 0; j < 4; ++j) {
      const unsigned short* wrow =
          W + (size_t)(ntile + j * 16 + lm) * 512 + t * 32 + half * 16;
      v16bf bfr = *(const v16bf*)wrow;       // one 32B load, used twice
      acc[0][j] = wmma_bf16(a0, bfr, acc[0][j]);
      acc[1][j] = wmma_bf16(a1, bfr, acc[1][j]);
    }
  }

#pragma unroll
  for (int i = 0; i < 2; ++i) {
#pragma unroll
    for (int j = 0; j < 4; ++j) {
      const int e = ntile + j * 16 + lm;
      const float bz = bias[e];
#pragma unroll
      for (int r = 0; r < 8; ++r) {
        const int gm = mtile + i * 16 + r + half * 8;  // C layout: M = r+8*half
        const unsigned short val = f2bf(acc[i][j][r] + bz);
        if (sel == 0) {
          Qb[(size_t)gm * 512 + e] = val;
        } else if (sel == 1) {
          Kb[(size_t)gm * 512 + e] = val;
        } else {
          const int bb = gm >> 11, nl = gm & 2047;
          VTb[((size_t)bb * 512 + e) * 2048 + nl] = val;
        }
      }
    }
  }
}

// ---------------------------------------------------------------------------
// Kernel 2: flash attention.  One wave owns 16 query rows x full D=512
// (acc[32] = 256 VGPRs; gfx1250 wave32 has a 1024-VGPR file).  Keys are
// streamed in blocks of 64 (4 S-tiles sharing each Q A-frag), online softmax
// rescale amortized over 64 keys, P staged C-layout -> A-layout through LDS.
// ---------------------------------------------------------------------------
__global__ __launch_bounds__(64) void flash_attn(
    const unsigned short* __restrict__ Qb, const unsigned short* __restrict__ Kb,
    const unsigned short* __restrict__ VTb, float* __restrict__ out) {
  __shared__ unsigned short pbuf[2][16 * 64];  // 2KB per wave
  const int lane = threadIdx.x & 31;
  const int wave = threadIdx.x >> 5;
  const int half = lane >> 4;
  const int lm   = lane & 15;
  const int qblk = blockIdx.x * 2 + wave;  // [0,128)
  const int b    = blockIdx.y;
  const int k8   = half * 8;
  unsigned short* pb = pbuf[wave];

  const unsigned short* Qrow = Qb + ((size_t)(b * 2048 + qblk * 16 + lm)) * 512;

  v8f acc[32];
#pragma unroll
  for (int j = 0; j < 32; ++j) acc[j] = vzero8();
  float m_r[8], l_r[8];
#pragma unroll
  for (int r = 0; r < 8; ++r) { m_r[r] = -3.0e38f; l_r[r] = 0.0f; }

  for (int kb = 0; kb < 2048; kb += 64) {
    // ---- S = Q . K^T for 64 keys (four 16x16 tiles), K-dim = D = 512 ----
    v8f s[4];
#pragma unroll
    for (int c = 0; c < 4; ++c) s[c] = vzero8();
    const unsigned short* Kbase =
        Kb + ((size_t)(b * 2048 + kb + lm)) * 512 + half * 16;
#pragma unroll 2
    for (int t = 0; t < 16; ++t) {
      v16bf qf = cat8(*(const v8bf*)(Qrow + t * 32 + 2 * k8),
                      *(const v8bf*)(Qrow + t * 32 + 2 * k8 + 8));
#pragma unroll
      for (int c = 0; c < 4; ++c) {
        v16bf bk = *(const v16bf*)(Kbase + (size_t)c * 16 * 512 + t * 32);
        s[c] = wmma_bf16(qf, bk, s[c]);
      }
    }

    // ---- online softmax over this 64-key block ----
#pragma unroll
    for (int r = 0; r < 8; ++r) {
      float a0 = s[0][r], a1 = s[1][r], a2 = s[2][r], a3 = s[3][r];
      float mx = fmaxf(fmaxf(a0, a1), fmaxf(a2, a3));
#pragma unroll
      for (int off = 1; off < 16; off <<= 1)
        mx = fmaxf(mx, __shfl_xor(mx, off, 32));  // stays within 16-lane half
      float mnew = fmaxf(m_r[r], mx);
      float corr = __expf(m_r[r] - mnew);
      m_r[r] = mnew;
      float p0 = __expf(a0 - mnew);
      float p1 = __expf(a1 - mnew);
      float p2 = __expf(a2 - mnew);
      float p3 = __expf(a3 - mnew);
      float ps = (p0 + p1) + (p2 + p3);
#pragma unroll
      for (int off = 1; off < 16; off <<= 1) ps += __shfl_xor(ps, off, 32);
      l_r[r] = l_r[r] * corr + ps;
#pragma unroll
      for (int j = 0; j < 32; ++j) acc[j][r] *= corr;
      const int row = r + half * 8;
      pb[row * 64 + lm]      = f2bf(p0);
      pb[row * 64 + 16 + lm] = f2bf(p1);
      pb[row * 64 + 32 + lm] = f2bf(p2);
      pb[row * 64 + 48 + lm] = f2bf(p3);
    }
    asm volatile("s_wait_dscnt 0" ::: "memory");

    // ---- reload P as two A-fragments (16x32 each) from LDS ----
    v16bf pf0 = cat8(*(const v8bf*)(pb + lm * 64 + 2 * k8),
                     *(const v8bf*)(pb + lm * 64 + 2 * k8 + 8));
    v16bf pf1 = cat8(*(const v8bf*)(pb + lm * 64 + 32 + 2 * k8),
                     *(const v8bf*)(pb + lm * 64 + 40 + 2 * k8));

    // ---- acc += P . V  (B-fragments contiguous from V^T rows) ----
#pragma unroll
    for (int j = 0; j < 32; ++j) {
      const unsigned short* vr =
          VTb + ((size_t)(b * 512 + j * 16 + lm)) * 2048 + kb + half * 16;
      acc[j] = wmma_bf16(pf0, *(const v16bf*)vr, acc[j]);
      acc[j] = wmma_bf16(pf1, *(const v16bf*)(vr + 32), acc[j]);
    }
  }

  // ---- epilogue: normalize by row sums and store f32 ----
#pragma unroll
  for (int r = 0; r < 8; ++r) {
    const float inv = 1.0f / l_r[r];
    const int row = qblk * 16 + r + half * 8;
    float* orow = out + ((size_t)(b * 2048 + row)) * 512 + lm;
#pragma unroll
    for (int j = 0; j < 32; ++j) orow[j * 16] = acc[j][r] * inv;
  }
}

// ---------------------------------------------------------------------------
extern "C" void kernel_launch(void* const* d_in, const int* in_sizes, int n_in,
                              void* d_out, int out_size, void* d_ws,
                              size_t ws_size, hipStream_t stream) {
  (void)in_sizes; (void)n_in; (void)out_size; (void)ws_size;
  const float* x  = (const float*)d_in[0];
  const float* Wq = (const float*)d_in[1];
  const float* bq = (const float*)d_in[2];
  const float* Wk = (const float*)d_in[3];
  const float* bk = (const float*)d_in[4];
  const float* Wv = (const float*)d_in[5];
  const float* bv = (const float*)d_in[6];

  unsigned char* ws = (unsigned char*)d_ws;
  unsigned short* xb  = (unsigned short*)(ws);
  unsigned short* Qb  = (unsigned short*)(ws + (size_t)16 * 1024 * 1024);
  unsigned short* Kb  = (unsigned short*)(ws + (size_t)32 * 1024 * 1024);
  unsigned short* VTb = (unsigned short*)(ws + (size_t)48 * 1024 * 1024);
  unsigned short* Wqb = (unsigned short*)(ws + (size_t)64 * 1024 * 1024);
  unsigned short* Wkb = (unsigned short*)(ws + (size_t)64 * 1024 * 1024 + 524288);
  unsigned short* Wvb = (unsigned short*)(ws + (size_t)64 * 1024 * 1024 + 1048576);

  const int nx = 8 * 2048 * 512;
  const int nw = 512 * 512;
  cvt_bf16<<<(nx + 255) / 256, 256, 0, stream>>>(x, xb, nx);
  cvt_bf16<<<(nw + 255) / 256, 256, 0, stream>>>(Wq, Wqb, nw);
  cvt_bf16<<<(nw + 255) / 256, 256, 0, stream>>>(Wk, Wkb, nw);
  cvt_bf16<<<(nw + 255) / 256, 256, 0, stream>>>(Wv, Wvb, nw);

  // Q, K, V^T projections with fused bias (sel = blockIdx.z)
  qkv_proj<<<dim3(128, 8, 3), 128, 0, stream>>>(xb, Wqb, Wkb, Wvb, bq, bk, bv,
                                                Qb, Kb, VTb);
  // Flash attention: grid (2048/16/2 qblock pairs, 8 batches), 2 waves/block
  flash_attn<<<dim3(64, 8), 64, 0, stream>>>(Qb, Kb, VTb, (float*)d_out);
}
